// our_adaptive_gnn_22119081575180
// MI455X (gfx1250) — compile-verified, matrix operand order checked by reading, not run on hardware
//
#include <hip/hip_runtime.h>
#include <hip/hip_bf16.h>
#include <math.h>

// Problem constants (match reference)
#define S_DIM  50000
#define E_DIM  20000
#define K_DIM  128
#define B_DIM  16384
#define NE_DIM 1000000

typedef __attribute__((ext_vector_type(16))) _Float16 v16h;
typedef __attribute__((ext_vector_type(8)))  _Float16 v8h;
typedef __attribute__((ext_vector_type(8)))  float    v8f;

__device__ __forceinline__ float sigmoidf_(float x) { return 1.0f / (1.0f + __expf(-x)); }

// Guaranteed-native non-returning f32 atomic add at device scope (L2 atomic
// units).  Avoids any chance of a CAS-loop fallback in the 10^9-atomic
// scatter phase; no return value -> STOREcnt-tracked, no VGPR writeback.
__device__ __forceinline__ void atomic_add_f32_dev(float* p, float v) {
    asm volatile("global_atomic_add_f32 %0, %1, off scope:SCOPE_DEV"
                 :
                 : "v"(p), "v"(v)
                 : "memory");
}

// ---------------------------------------------------------------------------
// PosLinear weight = 2*relu(-W)+W == |W|.  Pre-convert W1 (256x128) and
// W2 (128x256) to f16 |W| for the WMMA GEMMs. 32768 elements each.
// ---------------------------------------------------------------------------
__global__ void prep_weights(const float* __restrict__ W1, const float* __restrict__ W2,
                             _Float16* __restrict__ W1h, _Float16* __restrict__ W2h) {
    int t = blockIdx.x * blockDim.x + threadIdx.x;
    if (t < 32768) {
        W1h[t] = (_Float16)fabsf(W1[t]);
    } else {
        int u = t - 32768;
        if (u < 32768) W2h[u] = (_Float16)fabsf(W2[u]);
    }
}

// ---------------------------------------------------------------------------
// dst[r,:] = src[r,:] * (dA[r] + dB[r])   (diagonal term of each GCN layer;
// also serves as the zero-free initializer before edge scatter-adds)
// ---------------------------------------------------------------------------
__global__ void scale_rows(const float* __restrict__ src, const float* __restrict__ dA,
                           const float* __restrict__ dB, float* __restrict__ dst, int rows) {
    int t = blockIdx.x * blockDim.x + threadIdx.x;
    int n = rows * K_DIM;
    if (t < n) {
        int r = t >> 7;  // / K_DIM
        dst[t] = src[t] * (dA[r] + dB[r]);
    }
}

// ---------------------------------------------------------------------------
// SpMM scatter: dst[dIdx[e],:] += vals[e] * src[sIdx[e],:]
// One wave per edge iteration: 32 lanes x float4 = full 128-float row.
// Both src and dst matrices are L2-resident (<< 192MB), so the gathers and
// the native f32 atomic adds both resolve at L2 rate, not HBM rate.
// ---------------------------------------------------------------------------
__global__ void edge_scatter(const int* __restrict__ dIdx, const int* __restrict__ sIdx,
                             const float* __restrict__ vals, const float* __restrict__ src,
                             float* __restrict__ dst, int ne) {
    int wid  = (blockIdx.x * blockDim.x + threadIdx.x) >> 5;
    int lane = threadIdx.x & 31;
    int nw   = (gridDim.x * blockDim.x) >> 5;
    for (int e = wid; e < ne; e += nw) {
        int   d = dIdx[e];
        int   s = sIdx[e];
        float v = vals[e];
        const float4 g = *(const float4*)(src + (size_t)s * K_DIM + lane * 4);
        float* p = dst + (size_t)d * K_DIM + lane * 4;
        atomic_add_f32_dev(p + 0, v * g.x);
        atomic_add_f32_dev(p + 1, v * g.y);
        atomic_add_f32_dev(p + 2, v * g.z);
        atomic_add_f32_dev(p + 3, v * g.w);
    }
}

// ---------------------------------------------------------------------------
// x[b,k] = ikp[b,k] * (sigmoid(stat2[stu,k]+bias[stu]) - sigmoid(kdiff2[ex,k]))
//          * sigmoid(edisc[ex]);   stored f16 row-major for the WMMA GEMM.
// ---------------------------------------------------------------------------
__global__ void build_x(const int* __restrict__ stu, const int* __restrict__ ex,
                        const float* __restrict__ ikp, const float* __restrict__ stat2,
                        const float* __restrict__ bias, const float* __restrict__ kdiff2,
                        const float* __restrict__ edisc, _Float16* __restrict__ Xh) {
    int b = blockIdx.x;
    int k = threadIdx.x;
    int s = stu[b];
    int e = ex[b];
    float sb = sigmoidf_(stat2[(size_t)s * K_DIM + k] + bias[s]);
    float kd = sigmoidf_(kdiff2[(size_t)e * K_DIM + k]);
    float ed = sigmoidf_(edisc[e]);
    Xh[(size_t)b * K_DIM + k] = (_Float16)(ikp[(size_t)b * K_DIM + k] * (sb - kd) * ed);
}

// ---------------------------------------------------------------------------
// A/B fragment loader for V_WMMA_F32_16X16X32_F16 (ISA §7.12.2, 16-bit
// 16x32 A layout): lane (l, lo half) holds K {0..7}+{16..23}; hi half holds
// K {8..15}+{24..31}.  Both are two contiguous 8-half (16B) chunks.
// ---------------------------------------------------------------------------
__device__ __forceinline__ v16h load_frag(const _Float16* __restrict__ p, int hiOff) {
    v8h lo = *(const v8h*)(p + hiOff);
    v8h hi = *(const v8h*)(p + hiOff + 16);
    v16h a;
#pragma unroll
    for (int i = 0; i < 8; i++) { a[i] = lo[i]; a[i + 8] = hi[i]; }
    return a;
}

// ---------------------------------------------------------------------------
// out[M,N] = tanh(X[M,Kd] @ Wt[N,Kd]^T + bias[N])  via v_wmma_f32_16x16x32_f16.
// One wave per 16x16 output tile; f32 accumulation, Kd/32 WMMAs per tile.
// ---------------------------------------------------------------------------
template <bool HALF_OUT>
__global__ void gemm_wmma_tanh(const _Float16* __restrict__ X, const _Float16* __restrict__ Wt,
                               const float* __restrict__ bias, void* __restrict__ outv,
                               int M, int N, int Kd) {
    int wid  = (blockIdx.x * blockDim.x + threadIdx.x) >> 5;
    int lane = threadIdx.x & 31;
    int tilesN = N >> 4;
    int tm = (wid / tilesN) << 4;
    int tn = (wid % tilesN) << 4;
    if (tm >= M) return;
    int l     = lane & 15;
    int hiOff = (lane >= 16) ? 8 : 0;
    const _Float16* xr = X  + (size_t)(tm + l) * Kd;
    const _Float16* wr = Wt + (size_t)(tn + l) * Kd;
    v8f c = {};
    for (int kb = 0; kb < Kd; kb += 32) {
        v16h a = load_frag(xr + kb, hiOff);
        v16h b = load_frag(wr + kb, hiOff);
        // (neg_a, A, neg_b, B, c_mod, C, reuse_a, reuse_b)
        c = __builtin_amdgcn_wmma_f32_16x16x32_f16(false, a, false, b, (short)0, c, false, false);
    }
    // C/D layout: lanes 0-15 -> M = tm + r, lanes 16-31 -> M = tm + 8 + r; N = tn + (lane&15)
    int n     = tn + l;
    float bn  = bias[n];
    int mbase = tm + ((lane >= 16) ? 8 : 0);
#pragma unroll
    for (int r = 0; r < 8; r++) {
        float v = tanhf(c[r] + bn);
        if (HALF_OUT) ((_Float16*)outv)[(size_t)(mbase + r) * N + n] = (_Float16)v;
        else          ((float*)outv)[(size_t)(mbase + r) * N + n]    = v;
    }
}

// ---------------------------------------------------------------------------
// out[b] = sigmoid(sum_k |W3[k]| * H2[b,k] + b3).  One wave per row,
// float4 per lane + wave32 xor-shuffle reduction.
// ---------------------------------------------------------------------------
__global__ void final_dot(const float* __restrict__ H2, const float* __restrict__ W3,
                          const float* __restrict__ b3, float* __restrict__ out, int Bn) {
    int wid  = (blockIdx.x * blockDim.x + threadIdx.x) >> 5;
    int lane = threadIdx.x & 31;
    if (wid >= Bn) return;
    const float4 h = *(const float4*)(H2 + (size_t)wid * K_DIM + lane * 4);
    const float4 w = *(const float4*)(W3 + lane * 4);
    float s = fabsf(w.x) * h.x + fabsf(w.y) * h.y + fabsf(w.z) * h.z + fabsf(w.w) * h.w;
#pragma unroll
    for (int off = 16; off > 0; off >>= 1) s += __shfl_xor(s, off, 32);
    if (lane == 0) out[wid] = sigmoidf_(s + b3[0]);
}

// ---------------------------------------------------------------------------
extern "C" void kernel_launch(void* const* d_in, const int* in_sizes, int n_in,
                              void* d_out, int out_size, void* d_ws, size_t ws_size,
                              hipStream_t stream) {
    const int*   stu_id   = (const int*)d_in[0];
    const int*   input_ex = (const int*)d_in[1];
    const float* ikp      = (const float*)d_in[2];
    const int*   rows_1   = (const int*)d_in[3];
    const int*   cols_1   = (const int*)d_in[4];
    const int*   rows_0   = (const int*)d_in[5];
    const int*   cols_0   = (const int*)d_in[6];
    const float* v_ui_1   = (const float*)d_in[7];
    const float* v_iu_1   = (const float*)d_in[8];
    const float* v_ui_0   = (const float*)d_in[9];
    const float* v_iu_0   = (const float*)d_in[10];
    const float* d_i_1    = (const float*)d_in[11];
    const float* d_j_1    = (const float*)d_in[12];
    const float* d_i_0    = (const float*)d_in[13];
    const float* d_j_0    = (const float*)d_in[14];
    const float* stu_emb  = (const float*)d_in[15];
    const float* ex_emb   = (const float*)d_in[16];
    const float* stu_bias = (const float*)d_in[17];
    const float* edisc    = (const float*)d_in[18];
    const float* W1       = (const float*)d_in[19];
    const float* b1       = (const float*)d_in[20];
    const float* W2       = (const float*)d_in[21];
    const float* b2       = (const float*)d_in[22];
    const float* W3       = (const float*)d_in[23];
    const float* b3       = (const float*)d_in[24];

    // Workspace partition (~93 MB total)
    char*  ws  = (char*)d_ws;
    size_t off = 0;
    auto carve = [&](size_t bytes) -> char* {
        char* p = ws + off;
        off = (off + bytes + 255) & ~(size_t)255;
        return p;
    };
    float*    stat1  = (float*)carve((size_t)S_DIM * K_DIM * 4);  // 25.6 MB
    float*    kdiff1 = (float*)carve((size_t)E_DIM * K_DIM * 4);  // 10.2 MB
    float*    stat2  = (float*)carve((size_t)S_DIM * K_DIM * 4);  // 25.6 MB
    float*    kdiff2 = (float*)carve((size_t)E_DIM * K_DIM * 4);  // 10.2 MB
    _Float16* W1h    = (_Float16*)carve(256 * 128 * 2);
    _Float16* W2h    = (_Float16*)carve(128 * 256 * 2);
    _Float16* Xh     = (_Float16*)carve((size_t)B_DIM * 128 * 2); //  4.2 MB
    _Float16* H1h    = (_Float16*)carve((size_t)B_DIM * 256 * 2); //  8.4 MB
    float*    H2f    = (float*)carve((size_t)B_DIM * 128 * 4);    //  8.4 MB
    (void)ws_size; (void)in_sizes; (void)n_in; (void)out_size;

    const int TB = 256;
    const int scatterBlocks = 2048; // 16384 waves grid-striding 1M edges

    // Weight prep (independent of graph phase)
    prep_weights<<<(2 * 32768 + TB - 1) / TB, TB, 0, stream>>>(W1, W2, W1h, W2h);

    // ---- Graph layer 1: inputs are student_emb / exercise_emb ----
    scale_rows<<<(S_DIM * K_DIM + TB - 1) / TB, TB, 0, stream>>>(stu_emb, d_i_1, d_i_0, stat1, S_DIM);
    scale_rows<<<(E_DIM * K_DIM + TB - 1) / TB, TB, 0, stream>>>(ex_emb, d_j_1, d_j_0, kdiff1, E_DIM);
    edge_scatter<<<scatterBlocks, TB, 0, stream>>>(rows_1, cols_1, v_ui_1, ex_emb,  stat1,  NE_DIM);
    edge_scatter<<<scatterBlocks, TB, 0, stream>>>(rows_0, cols_0, v_ui_0, ex_emb,  stat1,  NE_DIM);
    edge_scatter<<<scatterBlocks, TB, 0, stream>>>(cols_1, rows_1, v_iu_1, stu_emb, kdiff1, NE_DIM);
    edge_scatter<<<scatterBlocks, TB, 0, stream>>>(cols_0, rows_0, v_iu_0, stu_emb, kdiff1, NE_DIM);

    // ---- Graph layer 2: inputs are stat1 / kdiff1 ----
    scale_rows<<<(S_DIM * K_DIM + TB - 1) / TB, TB, 0, stream>>>(stat1,  d_i_1, d_i_0, stat2,  S_DIM);
    scale_rows<<<(E_DIM * K_DIM + TB - 1) / TB, TB, 0, stream>>>(kdiff1, d_j_1, d_j_0, kdiff2, E_DIM);
    edge_scatter<<<scatterBlocks, TB, 0, stream>>>(rows_1, cols_1, v_ui_1, kdiff1, stat2,  NE_DIM);
    edge_scatter<<<scatterBlocks, TB, 0, stream>>>(rows_0, cols_0, v_ui_0, kdiff1, stat2,  NE_DIM);
    edge_scatter<<<scatterBlocks, TB, 0, stream>>>(cols_1, rows_1, v_iu_1, stat1,  kdiff2, NE_DIM);
    edge_scatter<<<scatterBlocks, TB, 0, stream>>>(cols_0, rows_0, v_iu_0, stat1,  kdiff2, NE_DIM);

    // ---- Batch feature assembly ----
    build_x<<<B_DIM, 128, 0, stream>>>(stu_id, input_ex, ikp, stat2, stu_bias, kdiff2, edisc, Xh);

    // ---- MLP head via WMMA (f16 in, f32 accumulate) ----
    // GEMM1: [16384,128] x [256,128]^T -> tanh -> f16 [16384,256]; 1024x16 tiles = 16384 waves
    gemm_wmma_tanh<true><<<16384 / 8, TB, 0, stream>>>(Xh, W1h, b1, (void*)H1h, B_DIM, 256, 128);
    // GEMM2: [16384,256] x [128,256]^T -> tanh -> f32 [16384,128]; 1024x8 tiles = 8192 waves
    gemm_wmma_tanh<false><<<8192 / 8, TB, 0, stream>>>(H1h, W2h, b2, (void*)H2f, B_DIM, 128, 256);
    // Final: dot with |W3| + sigmoid; one wave per output
    final_dot<<<B_DIM / 8, TB, 0, stream>>>(H2f, W3, b3, (float*)d_out, B_DIM);
}
